// LinearizedConv_16501264351610
// MI455X (gfx1250) — matrix operand order repaired and checked by reading, not compiled, not run on hardware
//
#include <hip/hip_runtime.h>
#include <hip/hip_bf16.h>
#include <math.h>
#include <stdint.h>

typedef __attribute__((ext_vector_type(16))) __bf16 v16bf;
typedef __attribute__((ext_vector_type(8)))  float  v8f;
typedef __attribute__((ext_vector_type(4)))  unsigned v4u;
typedef __attribute__((ext_vector_type(8)))  unsigned v8u;

#define N_NODE   4096
#define D_NODE   64
#define M_FEAT   128
#define B_BATCH  16
#define C_IN     32
#define T_LEN    12
#define HID      64
#define NT       (N_NODE * T_LEN)   /* 49152 positions per batch */
#define TC       (T_LEN * HID)      /* 768 */
#define EPSF        1e-4f
#define NORMALIZER  0.3535533905932738f   /* 64^-0.25 */
#define RATIO       0.08838834764831845f  /* 128^-0.5 */

/* workspace layout (bytes) ------------------------------------------------ */
#define WS_X      0u                        /* bf16 [NT][HID], reused per batch: 6 MB */
#define WS_KV     6291456u                  /* f32  [B][M][TC]: 6 MB                  */
#define WS_DASH2  (WS_KV + 6291456u)        /* f32  [N][M]: 2 MB                      */
#define WS_DIAG2  (WS_DASH2 + 2097152u)     /* f32  [N]                               */
#define WS_COLSUM (WS_DIAG2 + 16384u)       /* f32  [M]                               */
#define WS_DEN    (WS_COLSUM + 512u)        /* f32  [N]                               */
#define WS_GMAX   (WS_DEN + 16384u)         /* u32  [1]                               */
#define WS_V2T    (WS_GMAX + 256u)          /* bf16 [M][N] transposed v2p: 1 MB       */
/* total ~15.8 MiB */

#define OUT_V1  50331648   /* 16*64*4096*12 */
#define OUT_V2  50855936   /* OUT_V1 + 4096*128 */

/* monotonic float<->uint encoding so atomicMax(uint) == float max (order-indep) */
__device__ __forceinline__ unsigned f2mono(float f) {
  unsigned u = __float_as_uint(f);
  return (u & 0x80000000u) ? ~u : (u | 0x80000000u);
}
__device__ __forceinline__ float mono2f(unsigned u) {
  unsigned v = (u & 0x80000000u) ? (u & 0x7FFFFFFFu) : ~u;
  return __uint_as_float(v);
}

/* TDM: 2-D bf16 tile (tile_d0 x tile_d1 elements) global->LDS.
   D# group0/group1 packed per CDNA5 ISA 8.3/8.4; VADDR2/3 = NULL (2-D). */
__device__ __forceinline__ void tdm_load_tile_bf16(unsigned lds_off, const __bf16* gsrc,
                                                   unsigned tile_d0, unsigned tile_d1,
                                                   unsigned tensor_d0, unsigned tensor_d1,
                                                   unsigned stride0_elems) {
  unsigned long long ga = (unsigned long long)(uintptr_t)gsrc;
  v4u g0;
  g0[0] = 1u;                                    /* count=1, user descriptor        */
  g0[1] = lds_off;                               /* [63:32]  lds_addr               */
  g0[2] = (unsigned)ga;                          /* [95:64]  global_addr lo         */
  g0[3] = (unsigned)(ga >> 32) | (2u << 30);     /* [120:96] addr hi | type=2       */
  v8u g1;
  g1[0] = 1u << 16;                              /* data_size=1 (2 bytes), mask=0   */
  g1[1] = (tensor_d0 & 0xFFFFu) << 16;           /* tensor_dim0[15:0] @ bit48       */
  g1[2] = (tensor_d0 >> 16) | ((tensor_d1 & 0xFFFFu) << 16);
  g1[3] = (tensor_d1 >> 16) | (tile_d0 << 16);   /* tile_dim0 @ bit112              */
  g1[4] = tile_d1;                               /* tile_dim1; tile_dim2=0          */
  g1[5] = stride0_elems;                         /* tensor_dim0_stride lo           */
  g1[6] = 0u;
  g1[7] = 0u;
  asm volatile("tensor_load_to_lds %0, %1" :: "s"(g0), "s"(g1) : "memory");
}

/* ------------------------------------------------------------------------ */
__global__ void k_init(unsigned* gmax) {
  if (threadIdx.x == 0 && blockIdx.x == 0) *gmax = f2mono(-3.4e38f);
}

/* v1p: per-row stabilizer. one block per node, 128 threads (one per feature). */
__global__ void k_feat1(const float* __restrict__ nv1, const float* __restrict__ proj,
                        float* __restrict__ v1p) {
  __shared__ float red[128];
  int n = blockIdx.x, j = threadIdx.x;
  const float* row = nv1 + n * D_NODE;
  const float* pr  = proj + j * D_NODE;
  float dash = 0.f, sq = 0.f;
  for (int d = 0; d < D_NODE; ++d) { float r = row[d]; dash += r * pr[d]; sq += r * r; }
  dash *= NORMALIZER;
  float diag = 0.5f * sq * (NORMALIZER * NORMALIZER);
  red[j] = dash; __syncthreads();
  for (int s = 64; s > 0; s >>= 1) { if (j < s) red[j] = fmaxf(red[j], red[j + s]); __syncthreads(); }
  v1p[n * M_FEAT + j] = RATIO * (expf(dash - diag - red[0]) + EPSF);
}

/* v2p pass 1: dash2 + diag2 to ws, block max -> global atomicMax */
__global__ void k_feat2a(const float* __restrict__ nv2, const float* __restrict__ proj,
                         float* __restrict__ dash2, float* __restrict__ diag2,
                         unsigned* __restrict__ gmax) {
  __shared__ float red[128];
  int n = blockIdx.x, j = threadIdx.x;
  const float* row = nv2 + n * D_NODE;
  const float* pr  = proj + j * D_NODE;
  float dash = 0.f, sq = 0.f;
  for (int d = 0; d < D_NODE; ++d) { float r = row[d]; dash += r * pr[d]; sq += r * r; }
  dash *= NORMALIZER;
  dash2[n * M_FEAT + j] = dash;
  if (j == 0) diag2[n] = 0.5f * sq * (NORMALIZER * NORMALIZER);
  red[j] = dash; __syncthreads();
  for (int s = 64; s > 0; s >>= 1) { if (j < s) red[j] = fmaxf(red[j], red[j + s]); __syncthreads(); }
  if (j == 0) atomicMax(gmax, f2mono(red[0]));
}

/* v2p pass 2: exponentiate with the global stabilizer */
__global__ void k_feat2b(const float* __restrict__ dash2, const float* __restrict__ diag2,
                         const unsigned* __restrict__ gmax, float* __restrict__ v2p) {
  int i = blockIdx.x * blockDim.x + threadIdx.x;   /* 0..524287 */
  float stab = mono2f(*gmax);
  v2p[i] = RATIO * (expf(dash2[i] - diag2[i >> 7] - stab) + EPSF);
}

/* v2pT[m][n] = bf16(v2p[n][m]) : contiguous-K A operand for the kv GEMM */
__global__ void k_v2t(const float* __restrict__ v2p, __bf16* __restrict__ v2pT) {
  int i = blockIdx.x * 256 + threadIdx.x;          /* 0..524287 = m*4096+n */
  int mm = i >> 12, nn = i & 4095;
  v2pT[i] = (__bf16)v2p[nn * M_FEAT + mm];
}

/* colsum[j] = sum_n v2p[n,j] : one block per column */
__global__ void k_colsum(const float* __restrict__ v2p, float* __restrict__ colsum) {
  __shared__ float red[256];
  int j = blockIdx.x, t = threadIdx.x;
  float s = 0.f;
  for (int n = t; n < N_NODE; n += 256) s += v2p[n * M_FEAT + j];
  red[t] = s; __syncthreads();
  for (int k = 128; k > 0; k >>= 1) { if (t < k) red[t] += red[t + k]; __syncthreads(); }
  if (t == 0) colsum[j] = red[0];
}

/* den[n] = v1p[n,:] . colsum */
__global__ void k_den(const float* __restrict__ v1p, const float* __restrict__ colsum,
                      float* __restrict__ den) {
  int n = blockIdx.x * blockDim.x + threadIdx.x;
  float s = 0.f;
  for (int j = 0; j < M_FEAT; ++j) s += v1p[n * M_FEAT + j] * colsum[j];
  den[n] = s;
}

/* Gated 1x1 conv for one batch: x[p,hid] = sigmoid(Win@inp)*(Wout@inp), bf16 out.
   One WMMA (K=32 == C_IN) per 16x16 tile per gate. 4 waves cover hid=64. */
__global__ void k_conv(const float* __restrict__ inp_b,
                       const float* __restrict__ W_in, const float* __restrict__ b_in,
                       const float* __restrict__ W_out, const float* __restrict__ b_out,
                       __bf16* __restrict__ x_ws) {
  int wave = threadIdx.x >> 5, lane = threadIdx.x & 31;
  int hi = lane >> 4, lo = lane & 15;
  int koff = hi * 8;
  int mrow = wave * 16 + lo;
  v16bf a_in, a_out;
  #pragma unroll
  for (int e = 0; e < 16; ++e) {
    int k = hi * 16 + e;                                 /* contiguous K, matches B */
    a_in[e]  = (__bf16)W_in[mrow * C_IN + k];
    a_out[e] = (__bf16)W_out[mrow * C_IN + k];
  }
  int p = blockIdx.x * 16 + lo;
  int kbase = hi * 16;
  v16bf bt;
  #pragma unroll
  for (int e = 0; e < 16; ++e)
    bt[e] = (__bf16)inp_b[(kbase + e) * NT + p];         /* contiguous K, matches A */
  v8f zero = {};
  v8f hin  = __builtin_amdgcn_wmma_f32_16x16x32_bf16(false, a_in,  false, bt, (short)0, zero, false, false);
  v8f hout = __builtin_amdgcn_wmma_f32_16x16x32_bf16(false, a_out, false, bt, (short)0, zero, false, false);
  #pragma unroll
  for (int r = 0; r < 4; ++r) {
    int m0 = wave * 16 + hi * 8 + 2 * r;
    float xi0 = hin[2 * r]     + b_in[m0];
    float xo0 = hout[2 * r]    + b_out[m0];
    float xi1 = hin[2 * r + 1] + b_in[m0 + 1];
    float xo1 = hout[2 * r + 1]+ b_out[m0 + 1];
    unsigned short u0 = __builtin_bit_cast(unsigned short, (__bf16)(xo0 / (1.f + expf(-xi0))));
    unsigned short u1 = __builtin_bit_cast(unsigned short, (__bf16)(xo1 / (1.f + expf(-xi1))));
    *(unsigned*)&x_ws[p * HID + m0] = ((unsigned)u1 << 16) | u0;   /* packed 2xbf16 */
  }
}

/* kv[m,col] = sum_n v2p[n,m]*x[n,col]; K=4096 in 128 WMMA steps.
   8 waves cover m 0..127; x B-tile (32x16 bf16, 1KB) staged to LDS by the
   Tensor Data Mover (double-buffered), hand-off via TENSORcnt + barrier. */
__global__ void k_kv(const __bf16* __restrict__ v2pT, const __bf16* __restrict__ x_ws,
                     float* __restrict__ kv_b) {
  __shared__ __align__(16) __bf16 xt[2][32 * 16];
  int wave = threadIdx.x >> 5, lane = threadIdx.x & 31;
  int hi = lane >> 4, lo = lane & 15;
  int col0 = blockIdx.x * 16;
  int m = wave * 16 + lo;
  int koff = hi * 8, kbase0 = hi * 16;
  unsigned lds0 = (unsigned)(uintptr_t)(void*)&xt[0][0];   /* LDS aperture: low 32 bits = byte offset */
  unsigned lds1 = (unsigned)(uintptr_t)(void*)&xt[1][0];

  if (wave == 0) {
    tdm_load_tile_bf16(lds0, x_ws + col0, 16u, 32u, (unsigned)TC, 32u, (unsigned)TC);
    __builtin_amdgcn_s_wait_tensorcnt(0);
  }
  __syncthreads();

  v8f acc = {};
  int cur = 0;
  for (int kk = 0; kk < N_NODE; kk += 32) {
    if (wave == 0 && kk + 32 < N_NODE)
      tdm_load_tile_bf16(cur ? lds0 : lds1, x_ws + (size_t)(kk + 32) * TC + col0,
                         16u, 32u, (unsigned)TC, 32u, (unsigned)TC);
    v16bf at, bt;
    #pragma unroll
    for (int e = 0; e < 16; ++e) {
      int ka = kk + kbase0 + e;
      at[e] = v2pT[m * N_NODE + ka];                 /* contiguous 16 bf16 per lane */
      bt[e] = xt[cur][(kbase0 + e) * 16 + lo];
    }
    acc = __builtin_amdgcn_wmma_f32_16x16x32_bf16(false, at, false, bt, (short)0, acc, false, false);
    if (wave == 0) __builtin_amdgcn_s_wait_tensorcnt(0);
    __syncthreads();
    cur ^= 1;
  }
  #pragma unroll
  for (int r = 0; r < 8; ++r)
    kv_b[(wave * 16 + r + hi * 8) * TC + col0 + lo] = acc[r];
}

/* num = v1p @ kv, / den[n]; 512 threads, 16 waves compute one 16n x 768 super-
   tile; results transposed through LDS so the [b,c,n,t] writes are contiguous
   768B runs per channel. */
__global__ void k_num(const float* __restrict__ v1p, const float* __restrict__ kv_ws,
                      const float* __restrict__ den, float* __restrict__ out) {
  __shared__ float otile[HID * 192];                 /* 48 KB */
  int wave = threadIdx.x >> 5, lane = threadIdx.x & 31;
  int hi = lane >> 4, lo = lane & 15;
  int mtile = blockIdx.x & 255;
  int b = blockIdx.x >> 8;
  int nrow = mtile * 16 + lo;
  int koff = hi * 8, kbase0 = hi * 16;
  const float* kvb = kv_ws + b * M_FEAT * TC;

  v16bf at[4];                                      /* A frags reused across ntiles */
  #pragma unroll
  for (int kki = 0; kki < 4; ++kki)
    #pragma unroll
    for (int e = 0; e < 16; ++e) {
      int ka = kki * 32 + kbase0 + e;
      at[kki][e] = (__bf16)v1p[nrow * M_FEAT + ka];
    }

  for (int nt = wave; nt < 48; nt += 16) {
    int col = nt * 16 + lo;
    v8f acc = {};
    #pragma unroll
    for (int kki = 0; kki < 4; ++kki) {
      v16bf bt;
      #pragma unroll
      for (int e = 0; e < 16; ++e)
        bt[e] = (__bf16)kvb[(kki * 32 + kbase0 + e) * TC + col];
      acc = __builtin_amdgcn_wmma_f32_16x16x32_bf16(false, at[kki], false, bt, (short)0, acc, false, false);
    }
    int t = col >> 6, c = col & 63;
    #pragma unroll
    for (int r = 0; r < 8; ++r)
      otile[c * 192 + (r + hi * 8) * 12 + t] = acc[r];
  }
  __syncthreads();

  int n0 = mtile * 16;
  const float* dn = den + n0;
  float* ob = out + (size_t)b * HID * NT + (size_t)n0 * T_LEN;
  for (int i = threadIdx.x; i < HID * 192; i += 512) {
    int c = i / 192, r = i % 192;
    ob[(size_t)c * NT + r] = otile[i] / dn[r / 12];  /* 768B contiguous runs */
  }
}

/* ------------------------------------------------------------------------ */
extern "C" void kernel_launch(void* const* d_in, const int* in_sizes, int n_in,
                              void* d_out, int out_size, void* d_ws, size_t ws_size,
                              hipStream_t stream) {
  const float* input = (const float*)d_in[0];   /* [16,32,4096,12] */
  const float* nv1   = (const float*)d_in[1];   /* [4096,64] */
  const float* nv2   = (const float*)d_in[2];   /* [4096,64] */
  const float* W_in  = (const float*)d_in[3];   /* [64,32] */
  const float* b_in  = (const float*)d_in[4];   /* [64] */
  const float* W_out = (const float*)d_in[5];   /* [64,32] */
  const float* b_out = (const float*)d_in[6];   /* [64] */
  const float* proj  = (const float*)d_in[7];   /* [128,64] */

  float* out      = (float*)d_out;
  float* v1p_out  = out + OUT_V1;
  float* v2p_out  = out + OUT_V2;

  char* ws = (char*)d_ws;
  __bf16*   x_ws   = (__bf16*)(ws + WS_X);
  float*    kv_ws  = (float*)(ws + WS_KV);
  float*    dash2  = (float*)(ws + WS_DASH2);
  float*    diag2  = (float*)(ws + WS_DIAG2);
  float*    colsum = (float*)(ws + WS_COLSUM);
  float*    den    = (float*)(ws + WS_DEN);
  unsigned* gmax   = (unsigned*)(ws + WS_GMAX);
  __bf16*   v2pT   = (__bf16*)(ws + WS_V2T);

  k_init  <<<1,      256, 0, stream>>>(gmax);
  k_feat1 <<<N_NODE, 128, 0, stream>>>(nv1, proj, v1p_out);
  k_feat2a<<<N_NODE, 128, 0, stream>>>(nv2, proj, dash2, diag2, gmax);
  k_feat2b<<<2048,   256, 0, stream>>>(dash2, diag2, gmax, v2p_out);
  k_v2t   <<<2048,   256, 0, stream>>>(v2p_out, v2pT);
  k_colsum<<<M_FEAT, 256, 0, stream>>>(v2p_out, colsum);
  k_den   <<<16,     256, 0, stream>>>(v1p_out, colsum, den);

  for (int b = 0; b < B_BATCH; ++b) {
    const float* inp_b = input + (size_t)b * C_IN * NT;
    float*       kv_b  = kv_ws + (size_t)b * M_FEAT * TC;
    k_conv<<<NT / 16, 128, 0, stream>>>(inp_b, W_in, b_in, W_out, b_out, x_ws);
    k_kv  <<<TC / 16, 256, 0, stream>>>(v2pT, x_ws, kv_b);
  }

  k_num<<<B_BATCH * 256, 512, 0, stream>>>(v1p_out, kv_ws, den, out);
}
